// VotingEnsemble_91182155694590
// MI455X (gfx1250) — matrix operand order repaired and checked by reading, not compiled, not run on hardware
//
#include <hip/hip_runtime.h>

#define B_N 16384
#define D_K 512
#define C_N 1000
#define M_N 8

typedef __attribute__((ext_vector_type(16))) __bf16        v16bf;
typedef __attribute__((ext_vector_type(8)))  float         v8f;
typedef __attribute__((ext_vector_type(8)))  unsigned int  u32x8;

__device__ __forceinline__ unsigned int f2bf(float f) {
    // round-to-nearest-even f32 -> bf16 bits
    unsigned int u = __float_as_uint(f);
    return (u + 0x7FFFu + ((u >> 16) & 1u)) >> 16;
}
__device__ __forceinline__ float bf2f(unsigned int h) {
    return __uint_as_float(h << 16);
}
__device__ __forceinline__ v16bf mkv16(uint4 a, uint4 b) {
    u32x8 u;
    u[0] = a.x; u[1] = a.y; u[2] = a.z; u[3] = a.w;
    u[4] = b.x; u[5] = b.y; u[6] = b.z; u[7] = b.w;
    return __builtin_bit_cast(v16bf, u);
}

// ---------------------------------------------------------------- zero output
__global__ void zero_kernel(float* __restrict__ out, size_t n) {
    size_t i = (size_t)blockIdx.x * blockDim.x + threadIdx.x;
    if (i < n) out[i] = 0.0f;
}

// ------------- W [M,D,C] f32 -> Whi/Wlo [M,C,D] bf16 (transposed + split)
__global__ void wsplit_kernel(const float* __restrict__ W,
                              unsigned short* __restrict__ Whi,
                              unsigned short* __restrict__ Wlo) {
    size_t o = (size_t)blockIdx.x * blockDim.x + threadIdx.x;
    const size_t total = (size_t)M_N * C_N * D_K;
    if (o >= total) return;
    int d = (int)(o % D_K);
    int c = (int)((o / D_K) % C_N);
    int m = (int)(o / ((size_t)D_K * C_N));
    float v = W[((size_t)m * D_K + d) * C_N + c];
    unsigned int hi = f2bf(v);
    unsigned int lo = f2bf(v - bf2f(hi));
    Whi[o] = (unsigned short)hi;
    Wlo[o] = (unsigned short)lo;
}

// ------------- x [B,D] f32 -> Xhi/Xlo [B,D] bf16 (split only, same layout)
__global__ void xsplit_kernel(const float* __restrict__ x,
                              unsigned short* __restrict__ Xhi,
                              unsigned short* __restrict__ Xlo) {
    size_t o = (size_t)blockIdx.x * blockDim.x + threadIdx.x;
    const size_t total = (size_t)B_N * D_K;
    if (o >= total) return;
    float v = x[o];
    unsigned int hi = f2bf(v);
    unsigned int lo = f2bf(v - bf2f(hi));
    Xhi[o] = (unsigned short)hi;
    Xlo[o] = (unsigned short)lo;
}

// ------------------------------------------------------------- main GEMM+vote
__global__ __launch_bounds__(256, 2) void vote_kernel(
    const unsigned short* __restrict__ Xhi, const unsigned short* __restrict__ Xlo,
    const unsigned short* __restrict__ Whi, const unsigned short* __restrict__ Wlo,
    const float* __restrict__ bias, const float* __restrict__ coefs,
    float* __restrict__ out)
{
    const int tid  = threadIdx.x;
    const int wave = tid >> 5;
    const int lane = tid & 31;
    const int half = lane >> 4;     // 0: lanes 0-15, 1: lanes 16-31
    const int lid  = lane & 15;
    const int rt   = wave >> 2;     // 0..1 row tile
    const int cw   = wave & 3;      // 0..3 col group (64 cols each)

    const int rowblock = blockIdx.x;
    const int row0 = rowblock * 32 + rt * 16;
    const int arow = row0 + lid;                 // this lane's A-matrix row
    // loop-invariant 32-bit element offset: uniform base + voffset addressing
    const unsigned aoff = (unsigned)arow * D_K + half * 8;
    const float NEG_INF = -__builtin_inff();

    __shared__ float sv[32][4];
    __shared__ int   si[32][4];

    for (int m = 0; m < M_N; ++m) {
        // per-lane (in-lane) running best per row; cross-lane reduce deferred
        float runv[8];
        int   runi[8];
        #pragma unroll
        for (int i = 0; i < 8; ++i) { runv[i] = NEG_INF; runi[i] = 0; }

        for (int chunk = 0; chunk < 4; ++chunk) {
            const int colbase = chunk * 256 + cw * 64;
            v8f acc[4];
            #pragma unroll
            for (int t = 0; t < 4; ++t) acc[t] = (v8f){0,0,0,0,0,0,0,0};

            // per-tile 32-bit B offsets (clamped for the C tail)
            unsigned boff[4];
            #pragma unroll
            for (int t = 0; t < 4; ++t) {
                int c  = colbase + t * 16 + lid;
                int cc = c < C_N ? c : (C_N - 1);
                boff[t] = (unsigned)(m * C_N + cc) * D_K + half * 8;
            }

            #pragma unroll 2
            for (int k0 = 0; k0 < D_K; k0 += 32) {
                // ---- A operands (2x b128 each, already in WMMA lane layout)
                v16bf Ahi = mkv16(*(const uint4*)(Xhi + aoff + k0),
                                  *(const uint4*)(Xhi + aoff + k0 + 16));
                v16bf Alo = mkv16(*(const uint4*)(Xlo + aoff + k0),
                                  *(const uint4*)(Xlo + aoff + k0 + 16));

                // ---- B-hi operands for all 4 tiles
                v16bf Bh[4];
                #pragma unroll
                for (int t = 0; t < 4; ++t)
                    Bh[t] = mkv16(*(const uint4*)(Whi + boff[t] + k0),
                                  *(const uint4*)(Whi + boff[t] + k0 + 16));

                // term 1: hi*hi  (consecutive WMMAs hit different accumulators)
                #pragma unroll
                for (int t = 0; t < 4; ++t)
                    acc[t] = __builtin_amdgcn_wmma_f32_16x16x32_bf16(
                        false, Ahi, false, Bh[t], (short)0, acc[t], false, false);

                // ---- kick off all B-lo loads; they hide behind the next 4 WMMAs
                v16bf Bl[4];
                #pragma unroll
                for (int t = 0; t < 4; ++t)
                    Bl[t] = mkv16(*(const uint4*)(Wlo + boff[t] + k0),
                                  *(const uint4*)(Wlo + boff[t] + k0 + 16));

                // term 2: lo*hi  (reuses Bh)
                #pragma unroll
                for (int t = 0; t < 4; ++t)
                    acc[t] = __builtin_amdgcn_wmma_f32_16x16x32_bf16(
                        false, Alo, false, Bh[t], (short)0, acc[t], false, false);

                // term 3: hi*lo
                #pragma unroll
                for (int t = 0; t < 4; ++t)
                    acc[t] = __builtin_amdgcn_wmma_f32_16x16x32_bf16(
                        false, Ahi, false, Bl[t], (short)0, acc[t], false, false);
            }

            // ---- in-lane running argmax only (pure VALU, no cross-lane here).
            //      Each lane sees its columns in ascending order across t/chunk,
            //      so strict '>' keeps the first (lowest-index) maximum.
            #pragma unroll
            for (int t = 0; t < 4; ++t) {
                int c = colbase + t * 16 + lid;
                bool valid = c < C_N;
                int cb = valid ? c : (C_N - 1);
                float bb = bias[m * C_N + cb];      // clamped: no divergent branch
                #pragma unroll
                for (int i = 0; i < 8; ++i) {
                    float v = valid ? (acc[t][i] + bb) : NEG_INF;
                    if (v > runv[i]) { runv[i] = v; runi[i] = c; }
                }
            }
        }

        // ---- ONE cross-lane reduction per model (16-lane half groups)
        #pragma unroll
        for (int i = 0; i < 8; ++i) {
            float v = runv[i];
            int idx = runi[i];
            #pragma unroll
            for (int s = 8; s >= 1; s >>= 1) {
                float ov = __shfl_xor(v, s, 32);
                int   oi = __shfl_xor(idx, s, 32);
                if (ov > v || (ov == v && oi < idx)) { v = ov; idx = oi; }
            }
            runv[i] = v; runi[i] = idx;
        }

        // ---- merge the 4 col-groups per row, record vote (no atomics:
        //      one thread owns each row for the whole block / all models)
        __syncthreads();           // protect LDS against previous iteration's readers
        if (lid == 0) {
            #pragma unroll
            for (int i = 0; i < 8; ++i) {
                int r = rt * 16 + half * 8 + i;
                sv[r][cw] = runv[i];
                si[r][cw] = runi[i];
            }
        }
        __syncthreads();
        if (tid < 32) {
            float bv = sv[tid][0]; int bi = si[tid][0];
            #pragma unroll
            for (int q = 1; q < 4; ++q) {
                float v = sv[tid][q]; int ix = si[tid][q];
                if (v > bv || (v == bv && ix < bi)) { bv = v; bi = ix; }
            }
            int grow = rowblock * 32 + tid;
            out[(size_t)grow * C_N + bi] += coefs[m] * 0.125f;
        }
    }
}

// ----------------------------------------------------------------------------
extern "C" void kernel_launch(void* const* d_in, const int* in_sizes, int n_in,
                              void* d_out, int out_size, void* d_ws, size_t ws_size,
                              hipStream_t stream) {
    (void)in_sizes; (void)n_in; (void)out_size; (void)ws_size;
    const float* x     = (const float*)d_in[0];
    const float* W     = (const float*)d_in[1];
    const float* b     = (const float*)d_in[2];
    const float* coefs = (const float*)d_in[3];
    float* out         = (float*)d_out;

    // workspace layout (bf16 bit arrays):
    //   Whi: M*C*D (8.19 MB)   Wlo: M*C*D (8.19 MB)
    //   Xhi: B*D   (16.8 MB)   Xlo: B*D   (16.8 MB)     total ~50 MB
    const size_t wElems = (size_t)M_N * C_N * D_K;
    const size_t xElems = (size_t)B_N * D_K;
    unsigned short* Whi = (unsigned short*)d_ws;
    unsigned short* Wlo = Whi + wElems;
    unsigned short* Xhi = Wlo + wElems;
    unsigned short* Xlo = Xhi + xElems;

    size_t total_out = (size_t)B_N * C_N;
    zero_kernel<<<(unsigned)((total_out + 255) / 256), 256, 0, stream>>>(out, total_out);
    wsplit_kernel<<<(unsigned)((wElems + 255) / 256), 256, 0, stream>>>(W, Whi, Wlo);
    xsplit_kernel<<<(unsigned)((xElems + 255) / 256), 256, 0, stream>>>(x, Xhi, Xlo);

    vote_kernel<<<B_N / 32, 256, 0, stream>>>(Xhi, Xlo, Whi, Wlo, b, coefs, out);
}
